// GraphSAGE_438086664229
// MI455X (gfx1250) — matrix operand order repaired and checked by reading, not compiled, not run on hardware
//
#include <hip/hip_runtime.h>
#include <hip/hip_bf16.h>
#include <math.h>

typedef __attribute__((ext_vector_type(16))) __bf16 v16bf;
typedef __attribute__((ext_vector_type(8)))  float  v8f;
typedef __attribute__((ext_vector_type(2)))  __bf16 bf16x2;

#define DIM 128
#define GG  512
#define TT  10

// ---------------- degree / counts ----------------
__global__ void deg_kernel(const int* __restrict__ tgt, float* __restrict__ deg, int E) {
    int e = blockIdx.x * 256 + threadIdx.x;
    if (e < E) atomicAdd(&deg[tgt[e]], 1.0f);
}

__global__ void gcnt_kernel(const int* __restrict__ batch, float* __restrict__ gcnt, int N) {
    int n = blockIdx.x * 256 + threadIdx.x;
    if (n < N) atomicAdd(&gcnt[batch[n]], 1.0f);
}

// ---------------- edge scatter: agg[tgt] += h[src] ----------------
// one wave per edge; lane moves 4 contiguous floats (512B row = 32 lanes * float4)
__global__ void scatter_kernel(const float* __restrict__ h, const int* __restrict__ src,
                               const int* __restrict__ tgt, float* __restrict__ agg, int E) {
    int e    = blockIdx.x * 8 + (threadIdx.x >> 5);
    int lane = threadIdx.x & 31;
    if (e >= E) return;
    int s = src[e], t = tgt[e];
    float4 v = reinterpret_cast<const float4*>(h)[s * 32 + lane];
    float* dst = agg + t * DIM + lane * 4;
    atomicAdd(dst + 0, v.x);
    atomicAdd(dst + 1, v.y);
    atomicAdd(dst + 2, v.z);
    atomicAdd(dst + 3, v.w);
}

// ---------------- WMMA fragment loaders (bf16, 16x16x32) ----------------
// A 16x32: lanes 0-15 -> K {0..7,16..23}; lanes 16-31 -> K {8..15,24..31}; 2 halves/VGPR
__device__ __forceinline__ v16bf load_a_frag(const __bf16* sA, int m, int k0, int lane) {
    const unsigned* p = reinterpret_cast<const unsigned*>(sA);
    int khi = lane >> 4;
    v16bf f;
#pragma unroll
    for (int j = 0; j < 8; ++j) {
        int k = k0 + ((j < 4) ? (khi * 8 + 2 * j) : (16 + khi * 8 + 2 * (j - 4)));
        bf16x2 pr = __builtin_bit_cast(bf16x2, p[m * 64 + (k >> 1)]);
        f[2 * j] = pr.x; f[2 * j + 1] = pr.y;
    }
    return f;
}

// B 32x16: lane%16 = N col; lanes 0-15 K {0..7,16..23}, lanes 16-31 K {8..15,24..31}; sW is [n][k] stride 132
__device__ __forceinline__ v16bf load_b_frag(const __bf16* sW, int nl, int k0, int lane) {
    const unsigned* p = reinterpret_cast<const unsigned*>(sW);
    int kg = lane >> 4;
    v16bf f;
#pragma unroll
    for (int j = 0; j < 8; ++j) {
        int k = k0 + ((j < 4) ? (kg * 8 + 2 * j) : (16 + kg * 8 + 2 * (j - 4)));
        bf16x2 pr = __builtin_bit_cast(bf16x2, p[nl * 66 + (k >> 1)]);
        f[2 * j] = pr.x; f[2 * j + 1] = pr.y;
    }
    return f;
}

// ---------------- fused SAGE layer GEMM: h_out = (agg/deg)@Wl + bl + h_in@Wr ----------------
// block = 256 thr (8 waves); tile = 32 rows x 64 cols; gridDim = (ceil(N/32), 2)
__global__ void __launch_bounds__(256)
sage_gemm_kernel(const float* __restrict__ agg, const float* __restrict__ deg,
                 const float* __restrict__ h_in, const float* __restrict__ Wl,
                 const float* __restrict__ bl, const float* __restrict__ Wr,
                 float* __restrict__ h_out, int N) {
    __shared__ __bf16 sAagg[32 * DIM];   // [r][k]
    __shared__ __bf16 sAh  [32 * DIM];   // [r][k]
    __shared__ __bf16 sWl  [64 * 132];   // [n_local][k], padded stride 132
    __shared__ __bf16 sWr  [64 * 132];

    const int tid   = threadIdx.x;
    const int lane  = tid & 31;
    const int w     = tid >> 5;
    const int m0    = blockIdx.x * 32;
    const int nbase = blockIdx.y * 64;

    // stage transposed bf16 weight slices (coalesced global reads)
#pragma unroll 4
    for (int idx = tid; idx < 64 * DIM; idx += 256) {
        int k  = idx >> 6;
        int nl = idx & 63;
        sWl[nl * 132 + k] = (__bf16)Wl[k * DIM + nbase + nl];
        sWr[nl * 132 + k] = (__bf16)Wr[k * DIM + nbase + nl];
    }
    // stage activation tiles (agg pre-divided by clamped degree)
#pragma unroll 4
    for (int idx = tid; idx < 32 * DIM; idx += 256) {
        int r = idx >> 7;
        int c = idx & 127;
        int m = m0 + r; if (m >= N) m = N - 1;
        float d = deg[m]; d = d < 1.0f ? 1.0f : d;
        sAagg[idx] = (__bf16)(agg[m * DIM + c] * (1.0f / d));
        sAh[idx]   = (__bf16)h_in[m * DIM + c];
    }
    __syncthreads();

    const int r0    = (w >> 2) * 16;        // row sub-tile per wave group
    const int nloc0 = (w & 3) * 16;         // col sub-tile per wave
    const int mrow  = r0 + (lane & 15);
    const int ncol  = nloc0 + (lane & 15);

    v8f acc = {0.f, 0.f, 0.f, 0.f, 0.f, 0.f, 0.f, 0.f};
#pragma unroll
    for (int kt = 0; kt < 4; ++kt) {
        int k0 = kt * 32;
        v16bf aAgg = load_a_frag(sAagg, mrow, k0, lane);
        v16bf bWl  = load_b_frag(sWl, ncol, k0, lane);
        acc = __builtin_amdgcn_wmma_f32_16x16x32_bf16(false, aAgg, false, bWl,
                                                      (short)0, acc, false, false);
        v16bf aH  = load_a_frag(sAh, mrow, k0, lane);
        v16bf bWr = load_b_frag(sWr, ncol, k0, lane);
        acc = __builtin_amdgcn_wmma_f32_16x16x32_bf16(false, aH, false, bWr,
                                                      (short)0, acc, false, false);
    }

    const int n     = nbase + nloc0 + (lane & 15);
    const float bia = bl[n];
    const int mbase = m0 + r0 + (lane >> 4) * 8;   // C layout: VGPR r -> row r + 8*(lane/16)
#pragma unroll
    for (int r = 0; r < 8; ++r) {
        int m = mbase + r;
        if (m < N) h_out[m * DIM + n] = acc[r] + bia;
    }
}

// ---------------- global mean-pool accumulate ----------------
__global__ void pool_kernel(const float* __restrict__ h, const int* __restrict__ batch,
                            float* __restrict__ gsum, int layer_off, int N) {
    int idx = blockIdx.x * 256 + threadIdx.x;
    if (idx >= N * DIM) return;
    int n = idx >> 7;
    int d = idx & 127;
    atomicAdd(&gsum[batch[n] * (3 * DIM) + layer_off + d], h[idx]);
}

// ---------------- MLP head ----------------
__global__ void mlp1_kernel(const float* __restrict__ gsum, const float* __restrict__ gcnt,
                            const float* __restrict__ W1, const float* __restrict__ b1,
                            float* __restrict__ grelu) {
    int idx = blockIdx.x * 256 + threadIdx.x;   // G*128 threads
    if (idx >= GG * DIM) return;
    int g = idx >> 7;
    int j = idx & 127;
    float c = gcnt[g]; c = c < 1.0f ? 1.0f : c;
    float inv = 1.0f / c;
    float acc = 0.0f;
    const float* gs = gsum + g * (3 * DIM);
#pragma unroll 4
    for (int k = 0; k < 3 * DIM; ++k) acc += gs[k] * W1[k * DIM + j];
    acc = acc * inv + b1[j];
    grelu[idx] = acc > 0.0f ? acc : 0.0f;
}

__global__ void mlp2_kernel(const float* __restrict__ grelu, const float* __restrict__ W2,
                            const float* __restrict__ b2, float* __restrict__ out) {
    int idx = blockIdx.x * 256 + threadIdx.x;   // G*T threads
    if (idx >= GG * TT) return;
    int g = idx / TT;
    int t = idx - g * TT;
    float acc = b2[t];
    const float* gr = grelu + g * DIM;
#pragma unroll 4
    for (int k = 0; k < DIM; ++k) acc += gr[k] * W2[k * TT + t];
    out[idx] = 1.0f / (1.0f + expf(-acc));
}

// ---------------- host launcher ----------------
extern "C" void kernel_launch(void* const* d_in, const int* in_sizes, int n_in,
                              void* d_out, int out_size, void* d_ws, size_t ws_size,
                              hipStream_t stream) {
    (void)n_in; (void)out_size; (void)ws_size;
    const float* x     = (const float*)d_in[0];
    const int*   ei    = (const int*)d_in[1];
    const int*   batch = (const int*)d_in[2];
    const float* Wl    = (const float*)d_in[3];
    const float* bl    = (const float*)d_in[4];
    const float* Wr    = (const float*)d_in[5];
    const float* W1    = (const float*)d_in[6];
    const float* b1    = (const float*)d_in[7];
    const float* W2    = (const float*)d_in[8];
    const float* b2    = (const float*)d_in[9];

    const int E = in_sizes[1] / 2;
    const int N = in_sizes[2];
    const int* src = ei;
    const int* tgt = ei + E;

    // workspace layout (floats)
    float* ws    = (float*)d_ws;
    float* deg   = ws;                         // N
    float* gcnt  = deg + N;                    // G
    float* gsum  = gcnt + GG;                  // G*384
    float* grelu = gsum + (size_t)GG * 3 * DIM;// G*128
    float* agg   = grelu + (size_t)GG * DIM;   // N*128
    float* hA    = agg + (size_t)N * DIM;      // N*128
    float* hB    = hA  + (size_t)N * DIM;      // N*128

    // zero deg + gcnt + gsum + grelu in one contiguous shot
    size_t zero_floats = (size_t)N + GG + (size_t)GG * 3 * DIM + (size_t)GG * DIM;
    hipMemsetAsync(deg, 0, zero_floats * sizeof(float), stream);

    deg_kernel <<<(E + 255) / 256, 256, 0, stream>>>(tgt, deg, E);
    gcnt_kernel<<<(N + 255) / 256, 256, 0, stream>>>(batch, gcnt, N);

    const float* h_in = x;
    float* outs[3] = {hA, hB, hA};   // layer-2 may reuse hA (already pooled)
    for (int i = 0; i < 3; ++i) {
        hipMemsetAsync(agg, 0, (size_t)N * DIM * sizeof(float), stream);
        scatter_kernel<<<(E + 7) / 8, 256, 0, stream>>>(h_in, src, tgt, agg, E);
        dim3 grid((N + 31) / 32, 2);
        sage_gemm_kernel<<<grid, 256, 0, stream>>>(agg, deg, h_in,
                                                   Wl + (size_t)i * DIM * DIM,
                                                   bl + (size_t)i * DIM,
                                                   Wr + (size_t)i * DIM * DIM,
                                                   outs[i], N);
        pool_kernel<<<((size_t)N * DIM + 255) / 256, 256, 0, stream>>>(outs[i], batch, gsum,
                                                                       i * DIM, N);
        h_in = outs[i];
    }

    mlp1_kernel<<<(GG * DIM + 255) / 256, 256, 0, stream>>>(gsum, gcnt, W1, b1, grelu);
    mlp2_kernel<<<(GG * TT + 255) / 256, 256, 0, stream>>>(grelu, W2, b2, (float*)d_out);
}
